// MyFasterRCNN_45921790329161
// MI455X (gfx1250) — matrix operand order repaired and checked by reading, not compiled, not run on hardware
//
#include <hip/hip_runtime.h>
#include <hip/hip_bf16.h>

typedef __attribute__((ext_vector_type(16))) _Float16 v16h;
typedef __attribute__((ext_vector_type(8)))  _Float16 v8h;
typedef __attribute__((ext_vector_type(8)))  float    v8f;

// ---------------------------------------------------------------------------
// Images NCHW f32 -> NHWC f16
// ---------------------------------------------------------------------------
__global__ void img_to_nhwc_f16_kernel(const float* __restrict__ src,
                                       _Float16* __restrict__ dst,
                                       int Bn, int Cx, int H, int W) {
  int idx = blockIdx.x * blockDim.x + threadIdx.x;
  int total = Bn * Cx * H * W;
  if (idx >= total) return;
  int c = idx % Cx;
  int t = idx / Cx;
  int x = t % W; t /= W;
  int y = t % H;
  int b = t / H;
  dst[idx] = (_Float16)src[((size_t)(b * Cx + c) * H + y) * W + x];
}

// ---------------------------------------------------------------------------
// Pack conv weight OIHW f32 -> f16, WMMA B-fragment-major, PERMUTED K order:
//   k = (ky*3+kx)*Cin + ci   (spatial-major, channel-minor; matches NHWC A)
// ---------------------------------------------------------------------------
__global__ void pack_conv_w_frag_kernel(const float* __restrict__ w,
                                        _Float16* __restrict__ out,
                                        int Cin, int Cout, int Ktiles) {
  int idx = blockIdx.x * blockDim.x + threadIdx.x;
  int Nt = Cout >> 4;
  int total = Ktiles * Nt * 512;
  if (idx >= total) return;
  int i    = idx & 15;
  int lane = (idx >> 4) & 31;
  int rest = idx >> 9;
  int ct   = rest % Nt;
  int kt   = rest / Nt;
  int k  = kt * 32 + ((i >> 3) << 4) + ((lane >> 4) << 3) + (i & 7);
  int co = ct * 16 + (lane & 15);
  float v = 0.f;
  if (k < Cin * 9) {
    int r  = k / Cin;        // ky*3+kx
    int ci = k - r * Cin;
    v = w[(size_t)(co * Cin + ci) * 9 + r];
  }
  out[idx] = (_Float16)v;
}

// Fragment-major packing for a row-major K x N f32 matrix (K%32==0, N%16==0)
__global__ void pack_fc_w_frag_kernel(const float* __restrict__ w,
                                      _Float16* __restrict__ out,
                                      int K, int N) {
  int idx = blockIdx.x * blockDim.x + threadIdx.x;
  int Nt = N >> 4;
  int total = (K >> 5) * Nt * 512;
  if (idx >= total) return;
  int i    = idx & 15;
  int lane = (idx >> 4) & 31;
  int rest = idx >> 9;
  int ct   = rest % Nt;
  int kt   = rest / Nt;
  int k = kt * 32 + ((i >> 3) << 4) + ((lane >> 4) << 3) + (i & 7);
  int n = ct * 16 + (lane & 15);
  out[idx] = (_Float16)w[(size_t)k * N + n];
}

// ---------------------------------------------------------------------------
// Implicit-GEMM conv3x3 (pad=1), NHWC, f16 out. Two M-tiles per wave (MT=2)
// so each B fragment load feeds two WMMAs. VEC8: A-fragment half is one
// conditional v8h load (8 consecutive channels within one (ky,kx) tap).
// ---------------------------------------------------------------------------
template<int NT, bool VEC8>
__global__ void __launch_bounds__(128)
conv_wmma_kernel(const _Float16* __restrict__ act,
                 const _Float16* __restrict__ wpack,
                 _Float16* __restrict__ out,
                 int Cin, int cinShift, int H, int W, int Hout, int Wout,
                 int stride, int Kact, int Kp, int Cout, int Mpairs) {
  const int wave = threadIdx.x >> 5;
  const int lane = threadIdx.x & 31;
  const int pair = blockIdx.x * 4 + wave;
  if (pair >= Mpairs) return;
  const int ctBase = blockIdx.y * NT;
  const int Nt = Cout >> 4;

  const int HWo   = Hout * Wout;
  const int khalf = (lane >> 4) << 3;   // 0 or 8 (A frag)

  int iy0[2], ix0[2];
  size_t bBase[2];
#pragma unroll
  for (int t = 0; t < 2; ++t) {
    int mpix = pair * 32 + t * 16 + (lane & 15);
    int b  = mpix / HWo;
    int rm = mpix - b * HWo;
    int oy = rm / Wout;
    int ox = rm - oy * Wout;
    iy0[t] = oy * stride - 1;
    ix0[t] = ox * stride - 1;
    bBase[t] = (size_t)b * H * W * Cin;
  }

  const v8f vzero = {};
  const v8h z8 = {};
  v8f acc[2][NT];
#pragma unroll
  for (int t = 0; t < 2; ++t)
#pragma unroll
    for (int nt = 0; nt < NT; ++nt) acc[t][nt] = vzero;

  for (int k0 = 0; k0 < Kp; k0 += 32) {
    union AF { v16h v; v8h q[2]; _Float16 h[16]; } A[2];
    if (VEC8) {
#pragma unroll
      for (int half = 0; half < 2; ++half) {
        int kb  = k0 + half * 16 + khalf;       // multiple of 8
        int r   = kb >> cinShift;               // tap index ky*3+kx
        int ci0 = kb & (Cin - 1);
        int ky = r / 3;
        int kx = r - ky * 3;
#pragma unroll
        for (int t = 0; t < 2; ++t) {
          int iy = iy0[t] + ky, ix = ix0[t] + kx;
          v8h val = z8;
          if (iy >= 0 && iy < H && ix >= 0 && ix < W)
            val = *(const v8h*)(act + bBase[t] +
                                ((size_t)iy * W + ix) * Cin + ci0);
          A[t].q[half] = val;
        }
      }
    } else {
#pragma unroll
      for (int i = 0; i < 16; ++i) {
        int k = k0 + ((i >> 3) << 4) + khalf + (i & 7);
        _Float16 v0 = (_Float16)0.f, v1 = (_Float16)0.f;
        if (k < Kact) {
          int r  = k / Cin;
          int ci = k - r * Cin;
          int ky = r / 3;
          int kx = r - ky * 3;
          int iyA = iy0[0] + ky, ixA = ix0[0] + kx;
          int iyB = iy0[1] + ky, ixB = ix0[1] + kx;
          if (iyA >= 0 && iyA < H && ixA >= 0 && ixA < W)
            v0 = act[bBase[0] + ((size_t)iyA * W + ixA) * Cin + ci];
          if (iyB >= 0 && iyB < H && ixB >= 0 && ixB < W)
            v1 = act[bBase[1] + ((size_t)iyB * W + ixB) * Cin + ci];
        }
        A[0].h[i] = v0;
        A[1].h[i] = v1;
      }
    }
    const int kt = k0 >> 5;
#pragma unroll
    for (int nt = 0; nt < NT; ++nt) {
      const v16h bv = *(const v16h*)(wpack +
          ((((size_t)kt * Nt + (ctBase + nt)) * 32) + lane) * 16);
      acc[0][nt] = __builtin_amdgcn_wmma_f32_16x16x32_f16(
          false, A[0].v, false, bv, (short)0, acc[0][nt], false, false);
      acc[1][nt] = __builtin_amdgcn_wmma_f32_16x16x32_f16(
          false, A[1].v, false, bv, (short)0, acc[1][nt], false, false);
    }
  }

  // NHWC output is a row-major M x Cout matrix: out[p*Cout + co]
  const int rowOff = (lane >> 4) << 3;
#pragma unroll
  for (int t = 0; t < 2; ++t) {
#pragma unroll
    for (int nt = 0; nt < NT; ++nt) {
      const int co = (ctBase + nt) * 16 + (lane & 15);
#pragma unroll
      for (int r = 0; r < 8; ++r) {
        size_t p = (size_t)((pair * 2 + t) * 16 + rowOff + r);
        out[p * Cout + co] = (_Float16)acc[t][nt][r];
      }
    }
  }
}

// ---------------------------------------------------------------------------
// GroupNorm stats stage 1 (NHWC): grid (B*32, NCHUNK)
// ---------------------------------------------------------------------------
__global__ void gn_stats_partial_kernel(const _Float16* __restrict__ x,
                                        float* __restrict__ partial,
                                        int HW, int C, int Cpg) {
  const int bg = blockIdx.x;
  const int b  = bg >> 5;
  const int g  = bg & 31;
  const int ch = blockIdx.y;
  const _Float16* xp = x + (size_t)b * HW * C + g * Cpg;
  const int beg = (int)(((long)HW * ch) / gridDim.y);
  const int end = (int)(((long)HW * (ch + 1)) / gridDim.y);
  float s = 0.f, s2 = 0.f;
  for (int p = beg + threadIdx.x; p < end; p += blockDim.x) {
    const _Float16* row = xp + (size_t)p * C;
    for (int j = 0; j < Cpg; ++j) {
      float f = (float)row[j];
      s += f; s2 += f * f;
    }
  }
  __shared__ float sh[256], sh2[256];
  sh[threadIdx.x] = s; sh2[threadIdx.x] = s2;
  __syncthreads();
  for (int off = 128; off > 0; off >>= 1) {
    if ((int)threadIdx.x < off) {
      sh[threadIdx.x]  += sh[threadIdx.x + off];
      sh2[threadIdx.x] += sh2[threadIdx.x + off];
    }
    __syncthreads();
  }
  if (threadIdx.x == 0) {
    partial[(bg * gridDim.y + ch) * 2 + 0] = sh[0];
    partial[(bg * gridDim.y + ch) * 2 + 1] = sh2[0];
  }
}

// Stage 2: emit fused per-(b,c) scale/shift: y = x*scale + shift
__global__ void gn_finalize_kernel(const float* __restrict__ partial,
                                   const float* __restrict__ gw,
                                   const float* __restrict__ gb,
                                   float* __restrict__ scaleA,
                                   float* __restrict__ shiftA,
                                   int nchunks, float invN, int C, int Cpg,
                                   int BC) {
  int idx = blockIdx.x * blockDim.x + threadIdx.x;
  if (idx >= BC) return;
  int b = idx / C, c = idx - b * C;
  int bg = b * 32 + c / Cpg;
  float s = 0.f, s2 = 0.f;
  for (int ch = 0; ch < nchunks; ++ch) {
    s  += partial[(bg * nchunks + ch) * 2 + 0];
    s2 += partial[(bg * nchunks + ch) * 2 + 1];
  }
  float m   = s * invN;
  float var = s2 * invN - m * m;
  float rs  = rsqrtf(var + 1e-5f);
  float a   = rs * gw[c];
  scaleA[idx] = a;
  shiftA[idx] = gb[c] - m * a;
}

// Normalize+ReLU in place on NHWC f16; v8h spans 8 consecutive channels.
__global__ void gn_apply_kernel(_Float16* __restrict__ x,
                                const float* __restrict__ scaleA,
                                const float* __restrict__ shiftA,
                                size_t total8, int Cmask, size_t perBatch) {
  size_t i8 = (size_t)blockIdx.x * blockDim.x + threadIdx.x;
  if (i8 >= total8) return;
  size_t flat = i8 << 3;
  int c = (int)(flat & (size_t)Cmask);
  int b = (int)(flat / perBatch);
  const float* sA = scaleA + (size_t)b * (Cmask + 1) + c;
  const float* hA = shiftA + (size_t)b * (Cmask + 1) + c;
  v8h* p = (v8h*)x + i8;
  v8h v = *p;
  v8h o;
#pragma unroll
  for (int j = 0; j < 8; ++j) {
    float y = (float)v[j] * sA[j] + hA[j];
    o[j] = (_Float16)(y > 0.f ? y : 0.f);
  }
  *p = o;
}

// ---------------------------------------------------------------------------
// RoIAlign (NHWC feat): thread = (roi, pool-cell, 8-channel block)
// ---------------------------------------------------------------------------
__global__ void roi_align_nhwc_kernel(const _Float16* __restrict__ feat,
                                      const float* __restrict__ rois,
                                      _Float16* __restrict__ pooled,
                                      int R, int C, int Hf, int Wf) {
  int C8 = C >> 3;
  int idx = blockIdx.x * blockDim.x + threadIdx.x;
  int total = R * 49 * C8;
  if (idx >= total) return;
  int c8  = idx % C8;
  int tmp = idx / C8;
  int s   = tmp % 49;
  int r   = tmp / 49;
  int py = s / 7, px = s - py * 7;
  int c0 = c8 * 8;

  const float scale = 1.f / 16.f;
  int   b  = (int)rois[r * 5 + 0];
  float x1 = rois[r * 5 + 1] * scale - 0.5f;
  float y1 = rois[r * 5 + 2] * scale - 0.5f;
  float x2 = rois[r * 5 + 3] * scale - 0.5f;
  float y2 = rois[r * 5 + 4] * scale - 0.5f;
  float rw = x2 - x1, rh = y2 - y1;

  const _Float16* fb = feat + (size_t)b * Hf * Wf * C + c0;
  float acc[8];
#pragma unroll
  for (int j = 0; j < 8; ++j) acc[j] = 0.f;

#pragma unroll
  for (int sy = 0; sy < 2; ++sy) {
#pragma unroll
    for (int sx = 0; sx < 2; ++sx) {
      float ty = ((float)(py * 2 + sy) + 0.5f) / 14.f;
      float tx = ((float)(px * 2 + sx) + 0.5f) / 14.f;
      float yy = y1 + ty * rh;
      float xx = x1 + tx * rw;
      if (yy < -1.f || yy > (float)Hf || xx < -1.f || xx > (float)Wf) continue;
      float y = fminf(fmaxf(yy, 0.f), (float)(Hf - 1));
      float x = fminf(fmaxf(xx, 0.f), (float)(Wf - 1));
      int y0 = (int)floorf(y), x0 = (int)floorf(x);
      int y1i = min(y0 + 1, Hf - 1), x1i = min(x0 + 1, Wf - 1);
      float ly = y - (float)y0, lx = x - (float)x0;
      float w00 = (1.f - ly) * (1.f - lx);
      float w01 = (1.f - ly) * lx;
      float w10 = ly * (1.f - lx);
      float w11 = ly * lx;
      v8h t00 = *(const v8h*)(fb + ((size_t)y0 * Wf + x0)  * C);
      v8h t01 = *(const v8h*)(fb + ((size_t)y0 * Wf + x1i) * C);
      v8h t10 = *(const v8h*)(fb + ((size_t)y1i * Wf + x0)  * C);
      v8h t11 = *(const v8h*)(fb + ((size_t)y1i * Wf + x1i) * C);
#pragma unroll
      for (int j = 0; j < 8; ++j)
        acc[j] += w00 * (float)t00[j] + w01 * (float)t01[j]
                + w10 * (float)t10[j] + w11 * (float)t11[j];
    }
  }
  _Float16* po = pooled + (size_t)r * (C * 49) + s;
#pragma unroll
  for (int j = 0; j < 8; ++j)
    po[(size_t)(c0 + j) * 49] = (_Float16)(acc[j] * 0.25f);
}

// ---------------------------------------------------------------------------
// Dense GEMM via WMMA, two M-tiles per wave: C = relu(A @ B + bias).
// A row-major f16 (rows >= Mvalid zero), B fragment-major f16, f16 out.
// ---------------------------------------------------------------------------
template<int NT, bool RELU>
__global__ void __launch_bounds__(128)
gemm_wmma_kernel(const _Float16* __restrict__ A,
                 const _Float16* __restrict__ Bfrag,
                 const float* __restrict__ bias,
                 _Float16* __restrict__ Cc,
                 int Mvalid, int Mpairs, int N, int K) {
  const int wave = threadIdx.x >> 5;
  const int lane = threadIdx.x & 31;
  const int pair = blockIdx.x * 4 + wave;
  if (pair >= Mpairs) return;
  const int ctBase = blockIdx.y * NT;
  const int Nt = N >> 4;
  const int khalf = (lane >> 4) << 3;

  const int m0 = pair * 32 + (lane & 15);
  const int m1 = m0 + 16;
  const bool mok0 = (m0 < Mvalid);
  const bool mok1 = (m1 < Mvalid);
  const _Float16* Arow0 = A + (size_t)m0 * K;
  const _Float16* Arow1 = A + (size_t)m1 * K;

  const v8f vzero = {};
  const v8h z8 = {};
  v8f acc[2][NT];
#pragma unroll
  for (int t = 0; t < 2; ++t)
#pragma unroll
    for (int nt = 0; nt < NT; ++nt) acc[t][nt] = vzero;

  for (int k0 = 0; k0 < K; k0 += 32) {
    union AF { v16h v; v8h q[2]; } Af[2];
    if (mok0) {
      Af[0].q[0] = *(const v8h*)(Arow0 + k0 + khalf);
      Af[0].q[1] = *(const v8h*)(Arow0 + k0 + 16 + khalf);
    } else { Af[0].q[0] = z8; Af[0].q[1] = z8; }
    if (mok1) {
      Af[1].q[0] = *(const v8h*)(Arow1 + k0 + khalf);
      Af[1].q[1] = *(const v8h*)(Arow1 + k0 + 16 + khalf);
    } else { Af[1].q[0] = z8; Af[1].q[1] = z8; }
    const int kt = k0 >> 5;
#pragma unroll
    for (int nt = 0; nt < NT; ++nt) {
      const v16h bv = *(const v16h*)(Bfrag +
          ((((size_t)kt * Nt + (ctBase + nt)) * 32) + lane) * 16);
      acc[0][nt] = __builtin_amdgcn_wmma_f32_16x16x32_f16(
          false, Af[0].v, false, bv, (short)0, acc[0][nt], false, false);
      acc[1][nt] = __builtin_amdgcn_wmma_f32_16x16x32_f16(
          false, Af[1].v, false, bv, (short)0, acc[1][nt], false, false);
    }
  }

  const int rowOff = (lane >> 4) << 3;
#pragma unroll
  for (int t = 0; t < 2; ++t) {
#pragma unroll
    for (int nt = 0; nt < NT; ++nt) {
      int n = (ctBase + nt) * 16 + (lane & 15);
      float bv = bias[n];
#pragma unroll
      for (int r = 0; r < 8; ++r) {
        int p = (pair * 2 + t) * 16 + rowOff + r;
        float v = acc[t][nt][r] + bv;
        if (RELU) v = v > 0.f ? v : 0.f;
        Cc[(size_t)p * N + n] = (_Float16)v;
      }
    }
  }
}

// ---------------------------------------------------------------------------
// Heads + box decode
// ---------------------------------------------------------------------------
__global__ void heads_kernel(const _Float16* __restrict__ h,
                             const float* __restrict__ cw, const float* __restrict__ cb,
                             const float* __restrict__ bw, const float* __restrict__ bb,
                             float* __restrict__ scores, float* __restrict__ deltas,
                             int R, int HID) {
  int idx = blockIdx.x * blockDim.x + threadIdx.x;
  if (idx >= R * 6) return;
  int r = idx / 6, j = idx % 6;
  const v8h* h8 = (const v8h*)(h + (size_t)r * HID);
  float a;
  if (j < 2) {
    a = cb[j];
    for (int k8 = 0; k8 < HID / 8; ++k8) {
      v8h v = h8[k8];
#pragma unroll
      for (int t = 0; t < 8; ++t) a += (float)v[t] * cw[(k8 * 8 + t) * 2 + j];
    }
    scores[r * 2 + j] = a;
  } else {
    int jj = j - 2;
    a = bb[jj];
    for (int k8 = 0; k8 < HID / 8; ++k8) {
      v8h v = h8[k8];
#pragma unroll
      for (int t = 0; t < 8; ++t) a += (float)v[t] * bw[(k8 * 8 + t) * 4 + jj];
    }
    deltas[r * 4 + jj] = a;
  }
}

__global__ void decode_kernel(const float* __restrict__ rois,
                              const float* __restrict__ deltas,
                              float* __restrict__ boxes, int R) {
  int r = blockIdx.x * blockDim.x + threadIdx.x;
  if (r >= R) return;
  float bx1 = rois[r * 5 + 1], by1 = rois[r * 5 + 2];
  float bx2 = rois[r * 5 + 3], by2 = rois[r * 5 + 4];
  float w = bx2 - bx1, h = by2 - by1;
  float cx = bx1 + 0.5f * w, cy = by1 + 0.5f * h;
  float dx = deltas[r * 4 + 0], dy = deltas[r * 4 + 1];
  float dw = deltas[r * 4 + 2], dh = deltas[r * 4 + 3];
  float pcx = cx + dx * w, pcy = cy + dy * h;
  float pw = expf(dw) * w, ph = expf(dh) * h;
  boxes[r * 4 + 0] = pcx - 0.5f * pw;
  boxes[r * 4 + 1] = pcy - 0.5f * ph;
  boxes[r * 4 + 2] = pcx + 0.5f * pw;
  boxes[r * 4 + 3] = pcy + 0.5f * ph;
}

// ---------------------------------------------------------------------------
// Host orchestration
// ---------------------------------------------------------------------------
extern "C" void kernel_launch(void* const* d_in, const int* in_sizes, int n_in,
                              void* d_out, int out_size, void* d_ws, size_t ws_size,
                              hipStream_t stream) {
  (void)in_sizes; (void)n_in; (void)out_size; (void)ws_size;

  const float* images = (const float*)d_in[0];
  const float* rois   = (const float*)d_in[1];
  const float* conv_w[8]; const float* gn_w[8]; const float* gn_b[8];
  for (int i = 0; i < 8; ++i) {
    conv_w[i] = (const float*)d_in[2 + i];
    gn_w[i]   = (const float*)d_in[10 + i];
    gn_b[i]   = (const float*)d_in[18 + i];
  }
  const float* fc1_w = (const float*)d_in[26];
  const float* fc1_b = (const float*)d_in[27];
  const float* fc2_w = (const float*)d_in[28];
  const float* fc2_b = (const float*)d_in[29];
  const float* cls_w = (const float*)d_in[30];
  const float* cls_b = (const float*)d_in[31];
  const float* bbox_w = (const float*)d_in[32];
  const float* bbox_b = (const float*)d_in[33];
  float* out = (float*)d_out;

  char* base = (char*)d_ws;
  size_t off = 0;
  auto alloc = [&](size_t bytes) -> void* {
    off = (off + 255) & ~(size_t)255;
    void* p = base + off;
    off += bytes;
    return p;
  };

  const int Bn = 2, R = 600, HID = 1024;
  const int NCHUNK = 32;
  const int Ci[8] = {3, 64, 64, 128, 128, 256, 256, 256};
  const int Co[8] = {64, 64, 128, 128, 256, 256, 256, 256};
  const int St[8] = {2, 1, 2, 1, 2, 1, 2, 1};

  const size_t MAXACT = 32768000;  // 2*400*640*64 (largest activation)
  _Float16* actA    = (_Float16*)alloc(MAXACT * 2);
  _Float16* actB    = (_Float16*)alloc(MAXACT * 2);
  float*    partial = (float*)alloc((size_t)64 * NCHUNK * 2 * 4);
  float*    scaleA  = (float*)alloc((size_t)Bn * 256 * 4);
  float*    shiftA  = (float*)alloc((size_t)Bn * 256 * 4);
  _Float16* wpack[8]; int Kp[8];
  for (int i = 0; i < 8; ++i) {
    Kp[i] = ((Ci[i] * 9 + 31) / 32) * 32;
    wpack[i] = (_Float16*)alloc((size_t)Kp[i] * Co[i] * 2);
  }
  _Float16* fc1wh  = (_Float16*)alloc((size_t)12544 * 1024 * 2);
  _Float16* fc2wh  = (_Float16*)alloc((size_t)1024 * 1024 * 2);
  _Float16* pooled = (_Float16*)alloc((size_t)608 * 12544 * 2);
  _Float16* h1     = (_Float16*)alloc((size_t)608 * 1024 * 2);
  _Float16* h2     = (_Float16*)alloc((size_t)608 * 1024 * 2);
  float*    deltas = (float*)alloc((size_t)R * 4 * 4);

  // ---- weight prep + input convert/transpose ----
  for (int i = 0; i < 8; ++i) {
    int total = Kp[i] * Co[i];
    pack_conv_w_frag_kernel<<<(total + 255) / 256, 256, 0, stream>>>(
        conv_w[i], wpack[i], Ci[i], Co[i], Kp[i] / 32);
  }
  pack_fc_w_frag_kernel<<<((12544 / 32) * (1024 / 16) * 512 + 255) / 256, 256, 0,
                          stream>>>(fc1_w, fc1wh, 12544, 1024);
  pack_fc_w_frag_kernel<<<((1024 / 32) * (1024 / 16) * 512 + 255) / 256, 256, 0,
                          stream>>>(fc2_w, fc2wh, 1024, 1024);
  {
    int total = Bn * 3 * 800 * 1280;
    img_to_nhwc_f16_kernel<<<(total + 255) / 256, 256, 0, stream>>>(
        images, actA, Bn, 3, 800, 1280);
  }

  // ---- conv + GN + ReLU stack (NHWC f16 activations, in-place GN) ----
  _Float16* ain = actA;
  _Float16* aout = actB;
  int H = 800, W = 1280;
  for (int i = 0; i < 8; ++i) {
    int Ho = (H - 1) / St[i] + 1;
    int Wo = (W - 1) / St[i] + 1;
    int HWo = Ho * Wo;
    int Mpairs = Bn * HWo / 32;
    int Kact = Ci[i] * 9;
    int Cpg = Co[i] / 32;
    int cs  = (Ci[i] == 3) ? 0 : __builtin_ctz(Ci[i]);
    if (i == 0) {
      dim3 g((Mpairs + 3) / 4, 1);
      conv_wmma_kernel<4, false><<<g, 128, 0, stream>>>(
          ain, wpack[i], aout, Ci[i], cs, H, W, Ho, Wo, St[i],
          Kact, Kp[i], Co[i], Mpairs);
    } else if (Co[i] == 64) {
      dim3 g((Mpairs + 3) / 4, 1);
      conv_wmma_kernel<4, true><<<g, 128, 0, stream>>>(
          ain, wpack[i], aout, Ci[i], cs, H, W, Ho, Wo, St[i],
          Kact, Kp[i], Co[i], Mpairs);
    } else {
      dim3 g((Mpairs + 3) / 4, Co[i] / 128);
      conv_wmma_kernel<8, true><<<g, 128, 0, stream>>>(
          ain, wpack[i], aout, Ci[i], cs, H, W, Ho, Wo, St[i],
          Kact, Kp[i], Co[i], Mpairs);
    }
    gn_stats_partial_kernel<<<dim3(Bn * 32, NCHUNK), 256, 0, stream>>>(
        aout, partial, HWo, Co[i], Cpg);
    float invN = 1.f / (float)((size_t)Cpg * HWo);
    int BC = Bn * Co[i];
    gn_finalize_kernel<<<(BC + 255) / 256, 256, 0, stream>>>(
        partial, gn_w[i], gn_b[i], scaleA, shiftA, NCHUNK, invN, Co[i], Cpg, BC);
    size_t total8 = ((size_t)Bn * Co[i] * HWo) >> 3;
    gn_apply_kernel<<<(unsigned)((total8 + 255) / 256), 256, 0, stream>>>(
        aout, scaleA, shiftA, total8, Co[i] - 1, (size_t)HWo * Co[i]);
    _Float16* t = ain; ain = aout; aout = t;
    H = Ho; W = Wo;
  }
  // ain: final features NHWC (2, 50, 80, 256) f16

  // ---- RoIAlign -> pooled (608 x 12544 f16; reference (C,7,7) flattening) ----
  {
    int total = R * 49 * (256 / 8);
    roi_align_nhwc_kernel<<<(total + 255) / 256, 256, 0, stream>>>(
        ain, rois, pooled, R, 256, 50, 80);
  }

  // ---- FC head (M padded to 608 -> 19 tile pairs) ----
  {
    dim3 g1((19 + 3) / 4, 1024 / 128);
    gemm_wmma_kernel<8, true><<<g1, 128, 0, stream>>>(
        pooled, fc1wh, fc1_b, h1, R, 19, 1024, 12544);
    gemm_wmma_kernel<8, true><<<g1, 128, 0, stream>>>(
        h1, fc2wh, fc2_b, h2, R, 19, 1024, 1024);
  }

  // ---- cls/bbox heads + decode ----
  heads_kernel<<<(R * 6 + 255) / 256, 256, 0, stream>>>(
      h2, cls_w, cls_b, bbox_w, bbox_b, out, deltas, R, HID);
  decode_kernel<<<(R + 255) / 256, 256, 0, stream>>>(
      rois, deltas, out + 2 * R, R);
}
